// Attention_51041391346261
// MI455X (gfx1250) — matrix operand order repaired and checked by reading, not compiled, not run on hardware
//
#include <hip/hip_runtime.h>
#include <hip/hip_bf16.h>
#include <cstdint>
#include <cstddef>

#define BATCH  2
#define LSEQ   2048
#define DMODEL 2048
#define NH     16
#define HD     128

typedef __attribute__((ext_vector_type(16))) __bf16       bf16x16;
typedef __attribute__((ext_vector_type(8)))  float        f32x8;
typedef __attribute__((ext_vector_type(4)))  float        f32x4;
typedef __attribute__((ext_vector_type(4)))  unsigned int u32x4;
typedef __attribute__((ext_vector_type(2)))  unsigned int u32x2;

union FragAB { bf16x16 v; u32x4 q[2]; };

__device__ __forceinline__ unsigned short f2bf(float f) {
    unsigned int u = __builtin_bit_cast(unsigned int, f);
    u += 0x7FFFu + ((u >> 16) & 1u);          // round-to-nearest-even
    return (unsigned short)(u >> 16);
}

// ---------------- fp32 -> bf16 conversion (streaming) ----------------
__global__ void __launch_bounds__(256)
cvt_f32_bf16(const float* __restrict__ src, unsigned short* __restrict__ dst, int n4) {
    int i = blockIdx.x * blockDim.x + threadIdx.x;
    if (i >= n4) return;
    f32x4 v = ((const f32x4*)src)[i];
    u32x2 o;
    o.x = (unsigned int)f2bf(v.x) | ((unsigned int)f2bf(v.y) << 16);
    o.y = (unsigned int)f2bf(v.z) | ((unsigned int)f2bf(v.w) << 16);
    ((u32x2*)dst)[i] = o;
}

// ---------------- ALiBi bias tensor: (1,H,L,L) fp32 ----------------
__global__ void __launch_bounds__(256)
alibi_bias(float* __restrict__ out) {
    size_t idx = ((size_t)blockIdx.x * blockDim.x + threadIdx.x) * 4;
    int    j   = (int)(idx & (size_t)(LSEQ - 1));
    size_t t   = idx >> 11;
    int    i   = (int)(t & (size_t)(LSEQ - 1));
    int    h   = (int)(t >> 11);
    float slope = exp2f(-0.5f * (float)(h + 1));
    f32x4 o;
    o.x = -slope * fabsf((float)(j + 0 - i));
    o.y = -slope * fabsf((float)(j + 1 - i));
    o.z = -slope * fabsf((float)(j + 2 - i));
    o.w = -slope * fabsf((float)(j + 3 - i));
    *(f32x4*)(out + idx) = o;
}

// ---------------- WMMA GEMM: C = A(bf16, MxK) @ B(bf16, NxK)^T ----------------
// One wave computes a 16x64 tile. MODE 0: bf16 plain [MxN]. MODE 1: bf16 into
// V^T layout [B][H][HD][L]. MODE 2: fp32 plain [MxN].
template <int MODE>
__global__ void __launch_bounds__(32)
gemm_wmma(const unsigned short* __restrict__ A,
          const unsigned short* __restrict__ Bw,
          void* __restrict__ Out, int M, int N, int K) {
    const int lane  = threadIdx.x & 31;
    const int l15   = lane & 15;
    const int hh    = lane >> 4;
    const int nbase = blockIdx.x * 64;
    const int mbase = blockIdx.y * 16;

    f32x8 acc[4] = {};
    const unsigned short* arow = A + (size_t)(mbase + l15) * K;

    for (int k0 = 0; k0 < K; k0 += 32) {
        __builtin_prefetch(arow + k0 + 256, 0, 1);   // global_prefetch_b8
        FragAB a;
        a.q[0] = *(const u32x4*)(arow + k0 + hh * 16);
        a.q[1] = *(const u32x4*)(arow + k0 + hh * 16 + 8);
#pragma unroll
        for (int t = 0; t < 4; ++t) {
            const unsigned short* brow =
                Bw + (size_t)(nbase + t * 16 + l15) * K + k0 + hh * 16;
            FragAB b;
            b.q[0] = *(const u32x4*)(brow);
            b.q[1] = *(const u32x4*)(brow + 8);
            acc[t] = __builtin_amdgcn_wmma_f32_16x16x32_bf16(
                false, a.v, false, b.v, (short)0, acc[t], false, false);
        }
    }

    if (MODE == 0) {
        unsigned short* o = (unsigned short*)Out;
#pragma unroll
        for (int t = 0; t < 4; ++t) {
            int n = nbase + t * 16 + l15;
#pragma unroll
            for (int r = 0; r < 8; ++r)
                o[(size_t)(mbase + r + 8 * hh) * N + n] = f2bf(acc[t][r]);
        }
    } else if (MODE == 1) {
        unsigned short* o  = (unsigned short*)Out;
        int m0 = mbase + 8 * hh;
        int b  = m0 >> 11;               // / LSEQ
        int lq = m0 & (LSEQ - 1);
#pragma unroll
        for (int t = 0; t < 4; ++t) {
            int n = nbase + t * 16 + l15;
            int h = n >> 7, d = n & (HD - 1);
            u32x4 pk;
#pragma unroll
            for (int r = 0; r < 4; ++r)
                pk[r] = (unsigned int)f2bf(acc[t][2 * r]) |
                        ((unsigned int)f2bf(acc[t][2 * r + 1]) << 16);
            *(u32x4*)(o + ((size_t)((b * NH + h) * HD + d)) * LSEQ + lq) = pk;
        }
    } else {
        float* o = (float*)Out;
#pragma unroll
        for (int t = 0; t < 4; ++t) {
            int n = nbase + t * 16 + l15;
#pragma unroll
            for (int r = 0; r < 8; ++r)
                o[(size_t)(mbase + r + 8 * hh) * N + n] = acc[t][r];
        }
    }
}

// ---------------- Flash attention with ALiBi (one wave / 16 q-rows / head) ----
__global__ void __launch_bounds__(32)
attn_fa(const unsigned short* __restrict__ Qb,
        const unsigned short* __restrict__ Kb,
        const unsigned short* __restrict__ Vt,
        unsigned short* __restrict__ ctx) {
    __shared__ unsigned short sP[16 * 32];

    const int lane = threadIdx.x & 31;
    const int l15  = lane & 15;
    const int hh   = lane >> 4;
    const int q0   = blockIdx.x * 16;
    const int h    = blockIdx.y;
    const int b    = blockIdx.z;

    const float slope = exp2f(-0.5f * (float)(h + 1));
    const float scale = 0.0883883476483184f;   // 1/sqrt(128)

    // Q fragments, resident for the whole loop: A-layout 16x32 per k-step
    const unsigned short* qrow =
        Qb + ((size_t)(b * LSEQ) + q0 + l15) * DMODEL + h * HD;
    FragAB aq[4];
#pragma unroll
    for (int ks = 0; ks < 4; ++ks) {
        aq[ks].q[0] = *(const u32x4*)(qrow + ks * 32 + hh * 16);
        aq[ks].q[1] = *(const u32x4*)(qrow + ks * 32 + hh * 16 + 8);
    }

    f32x8 O[8] = {};
    float ms[8], ls[8];
#pragma unroll
    for (int r = 0; r < 8; ++r) { ms[r] = -3.0e38f; ls[r] = 0.0f; }

    for (int kb = 0; kb < LSEQ; kb += 32) {
        // ---- S = Q @ K^T for 32 keys (two 16-wide N tiles) ----
        f32x8 s[2];
#pragma unroll
        for (int nt = 0; nt < 2; ++nt) {
            f32x8 sa = {};
            const unsigned short* krow =
                Kb + ((size_t)(b * LSEQ) + kb + nt * 16 + l15) * DMODEL +
                h * HD + hh * 16;
#pragma unroll
            for (int ks = 0; ks < 4; ++ks) {
                FragAB bk;
                bk.q[0] = *(const u32x4*)(krow + ks * 32);
                bk.q[1] = *(const u32x4*)(krow + ks * 32 + 8);
                sa = __builtin_amdgcn_wmma_f32_16x16x32_bf16(
                    false, aq[ks].v, false, bk.v, (short)0, sa, false, false);
            }
            s[nt] = sa;
        }

        // ---- online softmax (fp32) with ALiBi bias ----
        int j0 = kb + l15;
#pragma unroll
        for (int r = 0; r < 8; ++r) {
            int   i  = q0 + r + 8 * hh;
            float v0 = s[0][r] * scale - slope * fabsf((float)(j0 - i));
            float v1 = s[1][r] * scale - slope * fabsf((float)(j0 + 16 - i));
            float t  = fmaxf(v0, v1);
            t = fmaxf(t, __shfl_xor(t, 1, 32));
            t = fmaxf(t, __shfl_xor(t, 2, 32));
            t = fmaxf(t, __shfl_xor(t, 4, 32));
            t = fmaxf(t, __shfl_xor(t, 8, 32));
            float mnew = fmaxf(ms[r], t);
            float corr = __expf(ms[r] - mnew);
            float p0   = __expf(v0 - mnew);
            float p1   = __expf(v1 - mnew);
            float u    = p0 + p1;
            u += __shfl_xor(u, 1, 32);
            u += __shfl_xor(u, 2, 32);
            u += __shfl_xor(u, 4, 32);
            u += __shfl_xor(u, 8, 32);
            ls[r] = ls[r] * corr + u;
            ms[r] = mnew;
#pragma unroll
            for (int t2 = 0; t2 < 8; ++t2) O[t2][r] *= corr;
            int Mr = r + 8 * hh;
            sP[Mr * 32 + l15]      = f2bf(p0);
            sP[Mr * 32 + 16 + l15] = f2bf(p1);
        }
        __syncthreads();
        // P (16x32) re-read in A-fragment layout
        FragAB pa;
        pa.q[0] = *(const u32x4*)(&sP[l15 * 32 + hh * 16]);
        pa.q[1] = *(const u32x4*)(&sP[l15 * 32 + hh * 16 + 8]);
        __syncthreads();

        // ---- O += P @ V  (V^T rows are contiguous) ----
        const unsigned short* vbase =
            Vt + ((size_t)((b * NH + h) * HD) + l15) * LSEQ + kb + hh * 16;
#pragma unroll
        for (int t2 = 0; t2 < 8; ++t2) {
            const unsigned short* vrow = vbase + (size_t)(t2 * 16) * LSEQ;
            FragAB bv;
            bv.q[0] = *(const u32x4*)(vrow);
            bv.q[1] = *(const u32x4*)(vrow + 8);
            O[t2] = __builtin_amdgcn_wmma_f32_16x16x32_bf16(
                false, pa.v, false, bv.v, (short)0, O[t2], false, false);
        }
    }

    // ---- epilogue: normalize and store bf16 context ----
#pragma unroll
    for (int r = 0; r < 8; ++r) {
        float  inv = 1.0f / ls[r];
        size_t row = ((size_t)(b * LSEQ) + q0 + r + 8 * hh) * DMODEL + h * HD;
#pragma unroll
        for (int t2 = 0; t2 < 8; ++t2)
            ctx[row + t2 * 16 + l15] = f2bf(O[t2][r] * inv);
    }
}

// ---------------------------------------------------------------------------
extern "C" void kernel_launch(void* const* d_in, const int* in_sizes, int n_in,
                              void* d_out, int out_size, void* d_ws, size_t ws_size,
                              hipStream_t stream) {
    (void)in_sizes; (void)n_in; (void)out_size; (void)ws_size;
    const float* x  = (const float*)d_in[0];
    const float* Wq = (const float*)d_in[1];
    const float* Wk = (const float*)d_in[2];
    const float* Wv = (const float*)d_in[3];
    const float* Wo = (const float*)d_in[4];
    float* out = (float*)d_out;

    const size_t nX = (size_t)BATCH * LSEQ * DMODEL;  // 8,388,608
    const size_t nW = (size_t)DMODEL * DMODEL;        // 4,194,304

    char* ws = (char*)d_ws;
    unsigned short* xb  = (unsigned short*)ws;  ws += nX * 2;
    unsigned short* wqb = (unsigned short*)ws;  ws += nW * 2;
    unsigned short* wkb = (unsigned short*)ws;  ws += nW * 2;
    unsigned short* wvb = (unsigned short*)ws;  ws += nW * 2;
    unsigned short* wob = (unsigned short*)ws;  ws += nW * 2;
    unsigned short* Qb  = (unsigned short*)ws;  ws += nX * 2;
    unsigned short* Kb  = (unsigned short*)ws;  ws += nX * 2;
    unsigned short* Vtb = (unsigned short*)ws;  ws += nX * 2;
    unsigned short* ctx = (unsigned short*)ws;  ws += nX * 2;

    // 1) bf16 conversions
    cvt_f32_bf16<<<(int)(nX / 4 / 256), 256, 0, stream>>>(x,  xb,  (int)(nX / 4));
    cvt_f32_bf16<<<(int)(nW / 4 / 256), 256, 0, stream>>>(Wq, wqb, (int)(nW / 4));
    cvt_f32_bf16<<<(int)(nW / 4 / 256), 256, 0, stream>>>(Wk, wkb, (int)(nW / 4));
    cvt_f32_bf16<<<(int)(nW / 4 / 256), 256, 0, stream>>>(Wv, wvb, (int)(nW / 4));
    cvt_f32_bf16<<<(int)(nW / 4 / 256), 256, 0, stream>>>(Wo, wob, (int)(nW / 4));

    // 2) projections
    dim3 g(DMODEL / 64, (BATCH * LSEQ) / 16), blk(32);
    gemm_wmma<0><<<g, blk, 0, stream>>>(xb, wqb, (void*)Qb,  BATCH * LSEQ, DMODEL, DMODEL);
    gemm_wmma<0><<<g, blk, 0, stream>>>(xb, wkb, (void*)Kb,  BATCH * LSEQ, DMODEL, DMODEL);
    gemm_wmma<1><<<g, blk, 0, stream>>>(xb, wvb, (void*)Vtb, BATCH * LSEQ, DMODEL, DMODEL);

    // 3) flash attention
    attn_fa<<<dim3(LSEQ / 16, NH, BATCH), blk, 0, stream>>>(Qb, Kb, Vtb, ctx);

    // 4) output projection (fp32 into d_out[0 .. nX))
    gemm_wmma<2><<<g, blk, 0, stream>>>(ctx, wob, (void*)out, BATCH * LSEQ, DMODEL, DMODEL);

    // 5) position_bias into d_out[nX ..)
    alibi_bias<<<(int)((size_t)NH * LSEQ * LSEQ / 4 / 256), 256, 0, stream>>>(out + nX);
}
